// GravNetLayer_13700945674819
// MI455X (gfx1250) — matrix operand order repaired and checked by reading, not compiled, not run on hardware
//
#include <hip/hip_runtime.h>
#include <stdint.h>

#define NPTS 32768
#define DIM  256
#define SDIM 4
#define PDIM 32
#define NEV  8
#define EPTS 4096
#define KNN  16
#define KA   320   // D + 2P

typedef __attribute__((ext_vector_type(16))) __bf16 v16bf;
typedef __attribute__((ext_vector_type(8)))  float  v8f;

__device__ __forceinline__ uint16_t f2bf(float f) {
  uint32_t u = __float_as_uint(f);
  return (uint16_t)((u + 0x7FFFu + ((u >> 16) & 1u)) >> 16);   // RNE
}

// two 16-byte chunks -> one 16 x bf16 fragment (32 bytes)
__device__ __forceinline__ v16bf load_frag(const uint16_t* p0, const uint16_t* p1) {
  union { uint4 u[2]; v16bf v; } c;
  c.u[0] = *(const uint4*)p0;
  c.u[1] = *(const uint4*)p1;
  return c.v;
}

// CDNA5 async global->LDS copy (16 bytes per lane), tracked by ASYNCcnt.
__device__ __forceinline__ void async_copy_b128(uint32_t lds_byte_addr, const void* gptr) {
  asm volatile("global_load_async_to_lds_b128 %0, %1, off"
               :: "v"(lds_byte_addr), "v"((unsigned long long)(uintptr_t)gptr)
               : "memory");
}
__device__ __forceinline__ void wait_asynccnt0() {
  asm volatile("s_wait_asynccnt 0x0" ::: "memory");
}
__device__ __forceinline__ uint32_t lds_addr32(const void* p) {
  return (uint32_t)(uintptr_t)p;   // LDS aperture base has zero low 32 bits
}

// ---------------- weight repack kernels ----------------
// WpT[48][256] bf16: rows 0..3 = Ws columns, 4..15 = 0, 16..47 = Wh columns
__global__ void prep_wpt(const float* __restrict__ Ws, const float* __restrict__ Wh,
                         uint16_t* __restrict__ WpT) {
  int idx = blockIdx.x * 256 + threadIdx.x;          // 48*256
  int n = idx >> 8, k = idx & 255;
  float v = 0.f;
  if (n < SDIM)       v = Ws[(size_t)k * SDIM + n];
  else if (n >= 16)   v = Wh[(size_t)k * PDIM + (n - 16)];
  WpT[(size_t)n * DIM + k] = f2bf(v);
}

// WoT[256][320] bf16: WoT[n][k] = Wo[k][n]
__global__ void prep_wot(const float* __restrict__ Wo, uint16_t* __restrict__ WoT) {
  int idx = blockIdx.x * 256 + threadIdx.x;          // 256*320
  int n = idx / KA, k = idx % KA;
  WoT[(size_t)n * KA + k] = f2bf(Wo[(size_t)k * DIM + n]);
}

// x f32 [N,256] -> bf16 into A[N,320] cols 0..255 (8 elems/thread, vectorized)
__global__ void cvt_x(const float* __restrict__ x, uint16_t* __restrict__ A) {
  int idx = blockIdx.x * 256 + threadIdx.x;          // N*256/8 threads
  int row = idx >> 5;
  int c8  = (idx & 31) << 3;
  const float4 a = *(const float4*)(x + (size_t)row * DIM + c8);
  const float4 b = *(const float4*)(x + (size_t)row * DIM + c8 + 4);
  uint4 o;
  o.x = (uint32_t)f2bf(a.x) | ((uint32_t)f2bf(a.y) << 16);
  o.y = (uint32_t)f2bf(a.z) | ((uint32_t)f2bf(a.w) << 16);
  o.z = (uint32_t)f2bf(b.x) | ((uint32_t)f2bf(b.y) << 16);
  o.w = (uint32_t)f2bf(b.z) | ((uint32_t)f2bf(b.w) << 16);
  *(uint4*)(A + (size_t)row * KA + c8) = o;
}

// ---------------- projection: s = xWs+bs, h = xWh+bh (WMMA bf16) ----------------
// WpT (24KB) staged once into LDS via async copies; B fragments read via DS.
__global__ __launch_bounds__(256) void gravnet_proj(
    const uint16_t* __restrict__ A, const uint16_t* __restrict__ WpT,
    const float* __restrict__ bs, const float* __restrict__ bh,
    float* __restrict__ s, float* __restrict__ h) {
  __shared__ __align__(16) uint16_t Ps[48 * DIM];    // 24 KB, layout == global

  const int lane = threadIdx.x & 31;
  const int wave = threadIdx.x >> 5;
  const int Rb   = blockIdx.x * 128 + wave * 16;     // 16-row M tile per wave
  const int nlo  = lane & 15;
  const int hi   = lane >> 4;
  const int arow = Rb + nlo;

  // stage all of WpT: 1536 chunks of 16B, 6 per thread
  const uint32_t psb = lds_addr32(Ps);
  #pragma unroll
  for (int q = 0; q < 6; ++q) {
    int c = threadIdx.x + (q << 8);
    async_copy_b128(psb + (uint32_t)c * 16u, WpT + (size_t)c * 8);
  }
  wait_asynccnt0();
  __syncthreads();

  v8f acc[3];
  #pragma unroll
  for (int t = 0; t < 3; ++t) acc[t] = (v8f){0,0,0,0,0,0,0,0};

  #pragma unroll
  for (int ks = 0; ks < DIM; ks += 32) {
    const uint16_t* pa = A + (size_t)arow * KA + ks + hi * 8;
    v16bf af = load_frag(pa, pa + 16);
    #pragma unroll
    for (int t = 0; t < 3; ++t) {
      const uint16_t* pb = &Ps[(t * 16 + nlo) * DIM + ks + hi * 8];
      v16bf bfr = load_frag(pb, pb + 16);
      acc[t] = __builtin_amdgcn_wmma_f32_16x16x32_bf16(
          false, af, false, bfr, (short)0, acc[t], false, false);
    }
  }

  const int rbase = Rb + hi * 8;
  if (nlo < SDIM) {                                   // tile 0 cols 0..3 -> s
    const float bsv = bs[nlo];
    #pragma unroll
    for (int r = 0; r < 8; ++r)
      s[(size_t)(rbase + r) * SDIM + nlo] = acc[0][r] + bsv;
  }
  #pragma unroll
  for (int t = 1; t < 3; ++t) {                       // tiles 1,2 -> h
    const int col = (t - 1) * 16 + nlo;
    const float bhv = bh[col];
    #pragma unroll
    for (int r = 0; r < 8; ++r)
      h[(size_t)(rbase + r) * PDIM + col] = acc[t][r] + bhv;
  }
}

// ---------------- kNN + weighted mean/max aggregation ----------------
__global__ __launch_bounds__(256) void gravnet_knn(
    const float* __restrict__ s, const float* __restrict__ h,
    uint16_t* __restrict__ A) {
  __shared__ float4 tileS[256];
  const int e = blockIdx.x >> 4;                      // 16 blocks per event
  const int pl = ((blockIdx.x & 15) << 8) + threadIdx.x;
  const int i  = e * EPTS + pl;
  const float4 si = ((const float4*)s)[i];

  float bd[KNN]; int bj[KNN];
  #pragma unroll
  for (int q = 0; q < KNN; ++q) { bd[q] = 3.4e38f; bj[q] = 0; }

  for (int t = 0; t < EPTS / 256; ++t) {
    __syncthreads();
    tileS[threadIdx.x] = ((const float4*)s)[e * EPTS + t * 256 + threadIdx.x];
    __syncthreads();
    #pragma unroll 4
    for (int j = 0; j < 256; ++j) {
      float4 sj = tileS[j];
      float dx = si.x - sj.x, dy = si.y - sj.y;
      float dz = si.z - sj.z, dw = si.w - sj.w;
      float d2 = dx * dx + dy * dy + dz * dz + dw * dw;
      if (d2 < bd[KNN - 1]) {                         // sorted-insert, static idx
        float dc = d2; int jc = t * 256 + j;
        #pragma unroll
        for (int q = 0; q < KNN; ++q) {
          if (dc < bd[q]) {
            float td = bd[q]; int tj = bj[q];
            bd[q] = dc; bj[q] = jc; dc = td; jc = tj;
          }
        }
      }
    }
  }

  float mean[PDIM], mx[PDIM];
  #pragma unroll
  for (int p = 0; p < PDIM; ++p) { mean[p] = 0.f; mx[p] = -3.4e38f; }
  #pragma unroll
  for (int k = 0; k < KNN; ++k) {
    const float wk = __expf(-10.f * bd[k]);
    const float* hj = h + (size_t)(e * EPTS + bj[k]) * PDIM;
    #pragma unroll
    for (int p = 0; p < PDIM; ++p) {
      float m = hj[p] * wk;
      mean[p] += m;
      mx[p] = fmaxf(mx[p], m);
    }
  }
  uint16_t* arow = A + (size_t)i * KA + DIM;
  #pragma unroll
  for (int p = 0; p < PDIM; ++p) {
    arow[p]        = f2bf(mean[p] * (1.f / KNN));
    arow[PDIM + p] = f2bf(mx[p]);
  }
}

// ---------------- output GEMM + bias + residual + LayerNorm (fused) ----------------
// Per k-step the 256x32 bf16 slice of WoT (16KB) is staged into LDS with async
// copies; all 8 waves then read B fragments from LDS instead of 8x from global.
__global__ __launch_bounds__(256) void gravnet_out(
    const uint16_t* __restrict__ A, const uint16_t* __restrict__ WoT,
    const float* __restrict__ x, const float* __restrict__ bo,
    const float* __restrict__ gamma, const float* __restrict__ beta,
    float* __restrict__ out) {
  __shared__ __align__(16) uint16_t Bs[256 * 32];    // 16 KB: row n at Bs[n*32]

  const int lane = threadIdx.x & 31;
  const int wave = threadIdx.x >> 5;
  const int Rb   = blockIdx.x * 128 + wave * 16;
  const int nlo  = lane & 15;
  const int hi   = lane >> 4;
  const int arow = Rb + nlo;
  const uint32_t bsb = lds_addr32(Bs);

  v8f acc[16];
  #pragma unroll
  for (int t = 0; t < 16; ++t) acc[t] = (v8f){0,0,0,0,0,0,0,0};

  for (int ks = 0; ks < KA; ks += 32) {
    // stage B slice: 1024 chunks of 16B (256 rows x 4 chunks), 4 per thread
    #pragma unroll
    for (int q = 0; q < 4; ++q) {
      int c = threadIdx.x + (q << 8);
      int n = c >> 2, sub = c & 3;
      async_copy_b128(bsb + (uint32_t)(n * 64 + sub * 16),
                      WoT + (size_t)n * KA + ks + sub * 8);
    }
    wait_asynccnt0();
    __syncthreads();

    const uint16_t* pa = A + (size_t)arow * KA + ks + hi * 8;
    v16bf af = load_frag(pa, pa + 16);
    #pragma unroll
    for (int t = 0; t < 16; ++t) {
      const uint16_t* pb = &Bs[(t * 16 + nlo) * 32 + hi * 8];
      v16bf bfr = load_frag(pb, pb + 16);
      acc[t] = __builtin_amdgcn_wmma_f32_16x16x32_bf16(
          false, af, false, bfr, (short)0, acc[t], false, false);
    }
    __syncthreads();
  }

  // bias + residual
  const int rbase = Rb + hi * 8;
  #pragma unroll
  for (int t = 0; t < 16; ++t) {
    const int col = t * 16 + nlo;
    const float bov = bo[col];
    #pragma unroll
    for (int r = 0; r < 8; ++r)
      acc[t][r] += bov + x[(size_t)(rbase + r) * DIM + col];
  }

  // LayerNorm: row sums live in one 16-lane half -> 4 shfl_xor steps
  float sum[8], ssq[8];
  #pragma unroll
  for (int r = 0; r < 8; ++r) { sum[r] = 0.f; ssq[r] = 0.f; }
  #pragma unroll
  for (int t = 0; t < 16; ++t)
    #pragma unroll
    for (int r = 0; r < 8; ++r) { sum[r] += acc[t][r]; ssq[r] += acc[t][r] * acc[t][r]; }
  #pragma unroll
  for (int m = 1; m <= 8; m <<= 1)
    #pragma unroll
    for (int r = 0; r < 8; ++r) {
      sum[r] += __shfl_xor(sum[r], m, 32);
      ssq[r] += __shfl_xor(ssq[r], m, 32);
    }
  float mu[8], rstd[8];
  #pragma unroll
  for (int r = 0; r < 8; ++r) {
    mu[r] = sum[r] * (1.f / DIM);
    float var = ssq[r] * (1.f / DIM) - mu[r] * mu[r];
    rstd[r] = rsqrtf(var + 1e-5f);
  }
  #pragma unroll
  for (int t = 0; t < 16; ++t) {
    const int col = t * 16 + nlo;
    const float g = gamma[col], b = beta[col];
    #pragma unroll
    for (int r = 0; r < 8; ++r)
      out[(size_t)(rbase + r) * DIM + col] = g * (acc[t][r] - mu[r]) * rstd[r] + b;
  }
}

// ---------------- host launch ----------------
extern "C" void kernel_launch(void* const* d_in, const int* in_sizes, int n_in,
                              void* d_out, int out_size, void* d_ws, size_t ws_size,
                              hipStream_t stream) {
  (void)in_sizes; (void)n_in; (void)out_size; (void)ws_size;
  const float* x     = (const float*)d_in[0];
  const float* Ws    = (const float*)d_in[2];
  const float* bs    = (const float*)d_in[3];
  const float* Wh    = (const float*)d_in[4];
  const float* bh    = (const float*)d_in[5];
  const float* Wo    = (const float*)d_in[6];
  const float* bo    = (const float*)d_in[7];
  const float* gamma = (const float*)d_in[8];
  const float* beta  = (const float*)d_in[9];
  float* out = (float*)d_out;

  // workspace layout (bytes)
  char* ws = (char*)d_ws;
  uint16_t* A   = (uint16_t*)(ws);                                   // 32768*320*2 = 20971520
  uint16_t* WpT = (uint16_t*)(ws + 20971520);                        // 48*256*2   = 24576
  uint16_t* WoT = (uint16_t*)(ws + 20971520 + 24576);                // 256*320*2  = 163840
  float*    s   = (float*)   (ws + 20971520 + 24576 + 163840);       // 32768*4*4  = 524288
  float*    h   = (float*)   (ws + 20971520 + 24576 + 163840 + 524288); // 32768*32*4

  prep_wpt<<<48, 256, 0, stream>>>(Ws, Wh, WpT);
  prep_wot<<<320, 256, 0, stream>>>(Wo, WoT);
  cvt_x<<<4096, 256, 0, stream>>>(x, A);
  gravnet_proj<<<NPTS / 128, 256, 0, stream>>>(A, WpT, bs, bh, s, h);
  gravnet_knn<<<NPTS / 256, 256, 0, stream>>>(s, h, A);
  gravnet_out<<<NPTS / 128, 256, 0, stream>>>(A, WoT, x, bo, gamma, beta, out);
}
